// SelfAttention_33062658244732
// MI455X (gfx1250) — matrix operand order repaired and checked by reading, not compiled, not run on hardware
//
#include <hip/hip_runtime.h>

typedef __attribute__((ext_vector_type(16))) __bf16 v16bf;
typedef __attribute__((ext_vector_type(8)))  __bf16 bf16x8;
typedef __attribute__((ext_vector_type(8)))  float  v8f;
typedef unsigned int v4u __attribute__((ext_vector_type(4)));
typedef int v8i __attribute__((ext_vector_type(8)));
typedef int v4i __attribute__((ext_vector_type(4)));

#define EDIM 2048
#define SEQ  2048
#define NBATCH 4
#define NH   16
#define HD   128
#define N3E  6144

// ---------------- fp32 -> bf16 convert ----------------
__global__ void cvt_f32_bf16(const float* __restrict__ s, __bf16* __restrict__ d, int n) {
    int i = blockIdx.x * blockDim.x + threadIdx.x;
    if (i < n) d[i] = (__bf16)s[i];
}

// A fragment (16x32 bf16, MxK): lane = row (lane%16); K chunks {kb..kb+7, kb+16..kb+23}
__device__ __forceinline__ v16bf load_fragA(const __bf16* p) {
    bf16x8 lo = *(const bf16x8*)p;
    bf16x8 hi = *(const bf16x8*)(p + 16);
    v16bf r;
#pragma unroll
    for (int i = 0; i < 8; ++i) { r[i] = lo[i]; r[i + 8] = hi[i]; }
    return r;
}
// B fragment (32x16 bf16, KxN): lane = col (lane%16); K chunks {kb..kb+7, kb+16..kb+23}
__device__ __forceinline__ v16bf load_fragB(const __bf16* p) {
    bf16x8 lo = *(const bf16x8*)p;
    bf16x8 hi = *(const bf16x8*)(p + 16);
    v16bf r;
#pragma unroll
    for (int i = 0; i < 8; ++i) { r[i] = lo[i]; r[i + 8] = hi[i]; }
    return r;
}

__device__ __forceinline__ float half16_max(float v) {
#pragma unroll
    for (int m = 1; m < 16; m <<= 1) v = fmaxf(v, __shfl_xor(v, m, 32));
    return v;
}
__device__ __forceinline__ float half16_sum(float v) {
#pragma unroll
    for (int m = 1; m < 16; m <<= 1) v += __shfl_xor(v, m, 32);
    return v;
}

// ---------------- TDM: 2D tile load (dword elements) to LDS ----------------
// D# per CDNA5 ISA: group0 = {count/lds_addr/global_addr/type},
// group1 = {data_size, tensor dims, tile dims, dim0 stride}. Tensor dims set huge (no OOB).
__device__ __forceinline__ void tdm_load_tile_2d(unsigned lds_off, const void* gptr,
                                                 unsigned tile_d0_dw, unsigned tile_d1,
                                                 unsigned stride_d0_dw) {
    unsigned long long ga = (unsigned long long)(uintptr_t)gptr;
    v4u g0;
    g0[0] = 1u;                                                  // count = 1 (valid D#)
    g0[1] = lds_off;                                             // LDS byte address
    g0[2] = (unsigned)ga;                                        // global_addr[31:0]
    g0[3] = (unsigned)((ga >> 32) & 0x01FFFFFFu) | (2u << 30);   // global_addr[56:32] | type=2
    const unsigned td0 = 0x40000000u, td1 = 0x40000000u;
    v8i g1;
    g1[0] = (int)(2u << 16);                                     // data_size = 4 bytes
    g1[1] = (int)((td0 & 0xFFFFu) << 16);                        // tensor_dim0[15:0]
    g1[2] = (int)((td0 >> 16) | ((td1 & 0xFFFFu) << 16));        // tensor_dim0[31:16] | tensor_dim1[15:0]
    g1[3] = (int)((td1 >> 16) | (tile_d0_dw << 16));             // tensor_dim1[31:16] | tile_dim0
    g1[4] = (int)tile_d1;                                        // tile_dim1 (tile_dim2=0)
    g1[5] = (int)stride_d0_dw;                                   // tensor_dim0_stride[31:0]
    g1[6] = 0;                                                   // stride hi | dim1_stride lo
    g1[7] = 0;
    v4i gz4 = {0, 0, 0, 0};
    v8i gz8 = {0, 0, 0, 0, 0, 0, 0, 0};
    __builtin_amdgcn_tensor_load_to_lds(g0, g1, gz4, gz4, gz8, 0);
}

// ---------------- QKV GEMM + bias + L2-normalize + layout ----------------
// grid (48, 64): x = N-tile (one (head,chunk) 128-col block), y = M-tile (128 tokens)
// block 256 threads = 8 waves; each wave computes 32(M) x 64(N) via 2x4 WMMA tiles.
__global__ __launch_bounds__(256)
void qkv_gemm_bf16(const __bf16* __restrict__ xb, const __bf16* __restrict__ wb,
                   const float* __restrict__ bias,
                   __bf16* __restrict__ qo, __bf16* __restrict__ ko, __bf16* __restrict__ vo) {
    __shared__ __bf16 sx[128 * 32];    // A tile [row][k]
    __shared__ __bf16 sw[128 * 32];    // B tile (W rows) [n][k]
    __shared__ __bf16 sy[128 * 128];   // result tile for row-norm pass
    __shared__ float  sscale[128];

    const int tid   = threadIdx.x;
    const int lane  = tid & 31;
    const int wv    = tid >> 5;
    const int n0    = blockIdx.x * 128;
    const int m0    = blockIdx.y * 128;
    const int waveM = (wv & 3) * 32;
    const int waveN = (wv >> 2) * 64;

    const int srow  = tid >> 1;
    const int shalf = tid & 1;

    const int lm  = lane & 15;
    const int kbA = (lane >> 4) * 8;
    const int kbB = (lane >> 4) * 8;

    const v8f vzero = {0.f, 0.f, 0.f, 0.f, 0.f, 0.f, 0.f, 0.f};
    v8f acc[2][4];
#pragma unroll
    for (int mt = 0; mt < 2; ++mt)
#pragma unroll
        for (int nt = 0; nt < 4; ++nt) acc[mt][nt] = vzero;

    for (int k0 = 0; k0 < EDIM; k0 += 32) {
        __syncthreads();
        {   // cooperative stage: each thread moves 32B of x-tile and 32B of W-tile
            const int4* s0 = (const int4*)(xb + (size_t)(m0 + srow) * EDIM + k0 + shalf * 16);
            int4* d0 = (int4*)(sx + srow * 32 + shalf * 16);
            d0[0] = s0[0]; d0[1] = s0[1];
            const int4* s1 = (const int4*)(wb + (size_t)(n0 + srow) * EDIM + k0 + shalf * 16);
            int4* d1 = (int4*)(sw + srow * 32 + shalf * 16);
            d1[0] = s1[0]; d1[1] = s1[1];
        }
        __syncthreads();
        v16bf af[2], bfg[4];
#pragma unroll
        for (int mt = 0; mt < 2; ++mt)
            af[mt] = load_fragA(sx + (waveM + mt * 16 + lm) * 32 + kbA);
#pragma unroll
        for (int nt = 0; nt < 4; ++nt)
            bfg[nt] = load_fragB(sw + (waveN + nt * 16 + lm) * 32 + kbB);
#pragma unroll
        for (int mt = 0; mt < 2; ++mt)
#pragma unroll
            for (int nt = 0; nt < 4; ++nt)
                acc[mt][nt] = __builtin_amdgcn_wmma_f32_16x16x32_bf16(
                    false, af[mt], false, bfg[nt], (short)0, acc[mt][nt], false, false);
    }

    __syncthreads();
    const int hidx = blockIdx.x / 3;  // head
    const int ck   = blockIdx.x % 3;  // 0=q 1=k 2=v
#pragma unroll
    for (int nt = 0; nt < 4; ++nt) {
        const int col = waveN + nt * 16 + lm;
        const float bv = bias[n0 + col];
#pragma unroll
        for (int mt = 0; mt < 2; ++mt)
#pragma unroll
            for (int r = 0; r < 8; ++r) {
                const int row = waveM + mt * 16 + ((lane < 16) ? r : r + 8);
                sy[row * 128 + col] = (__bf16)(acc[mt][nt][r] + bv);
            }
    }
    __syncthreads();
    if (tid < 128) {
        float s;
        if (ck < 2) {
            float ss = 0.f;
#pragma unroll 16
            for (int i = 0; i < 128; ++i) { float t = (float)sy[tid * 128 + i]; ss += t * t; }
            s = 1.0f / fmaxf(sqrtf(ss), 1e-12f);
        } else {
            s = 1.0f;
        }
        sscale[tid] = s;
    }
    __syncthreads();
    {
        const int row = tid >> 1;
        const int cb  = (tid & 1) * 64;
        const float sc = sscale[row];
        const int m  = m0 + row;
        const int bi = m >> 11;     // batch
        const int l  = m & 2047;    // seq pos
        if (ck == 2) {  // V: write transposed [B,H,HD,L]
            __bf16* dst = vo + ((size_t)(bi * NH + hidx) * HD) * SEQ + l;
#pragma unroll 8
            for (int i = 0; i < 64; ++i) dst[(size_t)(cb + i) * SEQ] = sy[row * 128 + cb + i];
        } else {        // Q/K normalized: [B,H,L,HD]
            __bf16* dst = (ck == 0 ? qo : ko) + ((size_t)(bi * NH + hidx) * SEQ + l) * HD + cb;
#pragma unroll 8
            for (int i = 0; i < 64; ++i)
                dst[i] = (__bf16)((float)sy[row * 128 + cb + i] * sc);
        }
    }
}

// ---------------- flash-style attention ----------------
// grid (SEQ/64, B*H); block 128 threads = 4 waves, 16 query rows/wave.
// K/V^T chunk staging via Tensor Data Mover (wave0: K, wave1: V^T).
__global__ __launch_bounds__(128)
void attn_fwd(const __bf16* __restrict__ qb, const __bf16* __restrict__ kb,
              const __bf16* __restrict__ vt, float* __restrict__ out) {
    extern __shared__ char smem[];
    __bf16* lds_k = (__bf16*)smem;        // [128 keys][128 hd]
    __bf16* lds_v = lds_k + 128 * 128;    // [128 hd][128 keys]  (v transposed)
    __bf16* lds_p = lds_v + 128 * 128;    // [4 waves][16 rows][128 keys]

    const int tid  = threadIdx.x;
    const int lane = tid & 31;
    const int wv   = tid >> 5;
    const int bh   = blockIdx.y;
    const int m0   = blockIdx.x * 64;
    const int qr0  = m0 + wv * 16;

    const int lm  = lane & 15;
    const int kbA = (lane >> 4) * 8;
    const int kbB = (lane >> 4) * 8;

    const __bf16* qbase = qb + (size_t)bh * SEQ * HD;
    const __bf16* kbase = kb + (size_t)bh * SEQ * HD;
    const __bf16* vbase = vt + (size_t)bh * HD * SEQ;

    // LDS byte offsets for TDM descriptors (generic LDS addr low 32 bits = LDS offset)
    const unsigned lds_k_off = (unsigned)(uintptr_t)lds_k;
    const unsigned lds_v_off = (unsigned)(uintptr_t)lds_v;

    v16bf qf[4];
#pragma unroll
    for (int ks = 0; ks < 4; ++ks)
        qf[ks] = load_fragA(qbase + (size_t)(qr0 + lm) * HD + ks * 32 + kbA);

    const v8f vzero = {0.f, 0.f, 0.f, 0.f, 0.f, 0.f, 0.f, 0.f};
    v8f oacc[8];
#pragma unroll
    for (int dn = 0; dn < 8; ++dn) oacc[dn] = vzero;
    float rm[8], rs[8];
#pragma unroll
    for (int r = 0; r < 8; ++r) { rm[r] = -INFINITY; rs[r] = 0.f; }

    for (int j0 = 0; j0 < SEQ; j0 += 128) {
        __syncthreads();  // previous-iteration LDS consumers done
        if (wv == 0) {
            // K chunk: 128 rows x 64 dwords, row stride 64 dwords (contiguous 32KB)
            tdm_load_tile_2d(lds_k_off, kbase + (size_t)j0 * HD, 64u, 128u, 64u);
            __builtin_amdgcn_s_wait_tensorcnt(0);
        } else if (wv == 1) {
            // V^T chunk: 128 rows (d) x 64 dwords, row stride 1024 dwords (SEQ*2B)
            tdm_load_tile_2d(lds_v_off, vbase + j0, 64u, 128u, 1024u);
            __builtin_amdgcn_s_wait_tensorcnt(0);
        }
        __syncthreads();  // publish TDM tiles to all waves

        // S = Q K^T  (16 rows x 128 keys per wave)
        v8f sc[8];
#pragma unroll
        for (int jn = 0; jn < 8; ++jn) {
            v8f c = vzero;
#pragma unroll
            for (int ks = 0; ks < 4; ++ks) {
                v16bf kf = load_fragB(lds_k + (jn * 16 + lm) * 128 + ks * 32 + kbB);
                c = __builtin_amdgcn_wmma_f32_16x16x32_bf16(
                        false, qf[ks], false, kf, (short)0, c, false, false);
            }
            sc[jn] = c;
        }

        // online softmax (C layout: vgpr r -> row r / r+8 by lane half, col = lane%16)
        float alpha[8];
        __bf16* pw = lds_p + wv * 16 * 128;
#pragma unroll
        for (int r = 0; r < 8; ++r) {
            float cm = sc[0][r];
#pragma unroll
            for (int jn = 1; jn < 8; ++jn) cm = fmaxf(cm, sc[jn][r]);
            cm = half16_max(cm);
            float m2 = fmaxf(rm[r], cm);
            alpha[r] = __expf(rm[r] - m2);
            const int prow = (lane < 16) ? r : r + 8;
            float rowsum = 0.f;
#pragma unroll
            for (int jn = 0; jn < 8; ++jn) {
                float pv = __expf(sc[jn][r] - m2);
                rowsum += pv;
                pw[prow * 128 + jn * 16 + lm] = (__bf16)pv;
            }
            rowsum = half16_sum(rowsum);
            rs[r] = rs[r] * alpha[r] + rowsum;
            rm[r] = m2;
        }
#pragma unroll
        for (int dn = 0; dn < 8; ++dn)
#pragma unroll
            for (int r = 0; r < 8; ++r) oacc[dn][r] = oacc[dn][r] * alpha[r];

        __syncthreads();  // P visible in LDS

        // O += P V  (A = P via LDS transpose, B = V^T rows)
#pragma unroll
        for (int ks = 0; ks < 4; ++ks) {
            v16bf pa = load_fragA(pw + lm * 128 + ks * 32 + kbA);
#pragma unroll
            for (int dn = 0; dn < 8; ++dn) {
                v16bf vf = load_fragB(lds_v + (dn * 16 + lm) * 128 + ks * 32 + kbB);
                oacc[dn] = __builtin_amdgcn_wmma_f32_16x16x32_bf16(
                               false, pa, false, vf, (short)0, oacc[dn], false, false);
            }
        }
    }

    const int b_ = bh >> 4, h_ = bh & 15;
#pragma unroll
    for (int dn = 0; dn < 8; ++dn)
#pragma unroll
        for (int r = 0; r < 8; ++r) {
            const int lrow = m0 + wv * 16 + ((lane < 16) ? r : r + 8);
            const int col  = dn * 16 + lm;
            out[((size_t)(b_ * SEQ + lrow)) * EDIM + h_ * HD + col] = oacc[dn][r] / rs[r];
        }
}

// ---------------- launcher ----------------
extern "C" void kernel_launch(void* const* d_in, const int* in_sizes, int n_in,
                              void* d_out, int out_size, void* d_ws, size_t ws_size,
                              hipStream_t stream) {
    const float* x = (const float*)d_in[0];
    const float* W = (const float*)d_in[1];
    const float* b = (const float*)d_in[2];
    float* out = (float*)d_out;

    // workspace layout (bf16): x(32MB) | W(24MB) | q(32MB) | k(32MB) | vT(32MB)
    __bf16* xb = (__bf16*)d_ws;
    __bf16* wb = xb + (size_t)NBATCH * SEQ * EDIM;
    __bf16* qo = wb + (size_t)N3E * EDIM;
    __bf16* ko = qo + (size_t)NBATCH * NH * SEQ * HD;
    __bf16* vo = ko + (size_t)NBATCH * NH * SEQ * HD;

    {
        int n = NBATCH * SEQ * EDIM;
        cvt_f32_bf16<<<(n + 255) / 256, 256, 0, stream>>>(x, xb, n);
    }
    {
        int n = N3E * EDIM;
        cvt_f32_bf16<<<(n + 255) / 256, 256, 0, stream>>>(W, wb, n);
    }
    qkv_gemm_bf16<<<dim3(48, 64), 256, 0, stream>>>(xb, wb, b, qo, ko, vo);

    size_t smem = (size_t)(128 * 128 + 128 * 128 + 4 * 16 * 128) * sizeof(__bf16);  // 80 KB
    attn_fwd<<<dim3(SEQ / 64, NBATCH * NH), 128, smem, stream>>>(qo, ko, vo, out);
}
